// CRF_58385785422115
// MI455X (gfx1250) — matrix operand order repaired and checked by reading, not compiled, not run on hardware
//
#include <hip/hip_runtime.h>

// CDNA5 / gfx1250 CRF forward algorithm.
// Step recurrence rewritten as f16 WMMA GEMM in the exp-domain:
//   alpha_t[i,b] = x_t[i,b] + m_b + log( sum_j exp(T[i,j]) * exp(alpha_{t-1}[j,b] - m_b) )
// A = exp(T) (constant, lives in VGPR fragments), B = exp(alpha - m) rebuilt each step in LDS.
// Hot-loop transcendentals use raw v_log_f32 / v_exp_f32 (safe: WMMA sum always in [1, 64e],
// exp argument always <= 0), avoiding the libm denorm/inf fixup chains on the critical path.

typedef __attribute__((ext_vector_type(16))) _Float16 v16h;
typedef __attribute__((ext_vector_type(8)))  float    v8f;

#define NCLS 64
#define TLEN 512
#define EA_STRIDE 72   // halves per row: 64 + 8 pad -> conflict-free banks, keeps 16B align

#define LOG2E 1.4426950408889634f
#define LN2   0.6931471805599453f

__global__ __launch_bounds__(128) void crf_forward_wmma(
    const float* __restrict__ pad_x,        // [B, T, 64]
    const float* __restrict__ trans,        // [64, 64]
    const float* __restrict__ orig,         // [64]
    const int*   __restrict__ batch_sizes,  // [B]
    float*       __restrict__ ws_norm)      // [B] out: normalizer per batch
{
    __shared__ _Float16 ealphaT[16 * EA_STRIDE];  // [b_local][class] f16 exp(alpha - m)
    __shared__ float    colmax[4 * 32];           // per-wave per-lane partial column maxima
    __shared__ float    norm_part[8 * 16];        // [wave*2+half][b_local] normalizer parts

    const int tid      = threadIdx.x;
    const int w        = tid >> 5;        // wave 0..3 -> output classes [16w, 16w+16)
    const int lane     = tid & 31;
    const int half_sel = lane >> 4;       // 0: lanes 0-15, 1: lanes 16-31
    const int bl       = lane & 15;       // local batch column (WMMA N index)
    const int b0       = blockIdx.x * 16;
    const int M0       = w * 16 + half_sel * 8;  // first of this lane's 8 D-rows

    norm_part[tid] = 0.0f;  // blockDim == 128 == 8*16 slots, one per thread

    // ---- per-lane batch size + workgroup max step count ----
    const int bs = batch_sizes[b0 + bl];
    int Tmax = 1;
    for (int k = 0; k < 16; ++k) {
        int v = batch_sizes[b0 + k];
        Tmax = v > Tmax ? v : Tmax;
    }

    // ---- A fragments: exp(transition), constant across all steps ----
    // 16-bit A 16x32 layout: lanes 0-15: halves 0..7 = K0..7, 8..15 = K16..23
    //                        lanes16-31: halves 0..7 = K8..15, 8..15 = K24..31
    const int arow  = w * 16 + bl;        // A-matrix M = lane&15 for both half-waves
    const int kbase = half_sel * 8;
    union { v16h v; _Float16 h[16]; } a0u, a1u;
    #pragma unroll
    for (int h = 0; h < 16; ++h) {
        int k = (h < 8) ? (kbase + h) : (16 + kbase + (h - 8));
        a0u.h[h] = (_Float16)__expf(trans[arow * NCLS + k]);        // K chunk 0..31
        a1u.h[h] = (_Float16)__expf(trans[arow * NCLS + k + 32]);   // K chunk 32..63
    }

    // origination scores for this lane's 8 classes (used at t==0)
    float og[8];
    #pragma unroll
    for (int r = 0; r < 8; ++r) og[r] = orig[M0 + r];

    float mprev = 0.0f;  // column max used when building previous step's e_alpha
    const float* xbase = pad_x + (size_t)(b0 + bl) * TLEN * NCLS + M0;

    __syncthreads();

    for (int t = 0; t < Tmax; ++t) {
        // x_t for this lane's 8 classes in its column (8 consecutive floats)
        const float* xp = xbase + (size_t)t * NCLS;
        float4 x0 = *(const float4*)(xp);
        float4 x1 = *(const float4*)(xp + 4);
        if (t + 1 < Tmax) __builtin_prefetch(xp + NCLS, 0, 1);  // global_prefetch_b8

        float aval[8];
        if (t == 0) {
            aval[0] = x0.x + og[0]; aval[1] = x0.y + og[1];
            aval[2] = x0.z + og[2]; aval[3] = x0.w + og[3];
            aval[4] = x1.x + og[4]; aval[5] = x1.y + og[5];
            aval[6] = x1.z + og[6]; aval[7] = x1.w + og[7];
        } else {
            // B fragments (e_alpha of previous step) from LDS.
            // 16-bit B layout: lanes 0-15 halves = K0..7, K16..23; lanes 16-31 halves = K8..15, K24..31, N = lane&15
            const int kb = half_sel * 8;
            union { v16h v; uint4 u[2]; } bu0, bu1;
            const uint4* s0 = (const uint4*)&ealphaT[bl * EA_STRIDE + kb];
            bu0.u[0] = s0[0];
            bu0.u[1] = s0[2];
            const uint4* s1 = (const uint4*)&ealphaT[bl * EA_STRIDE + kb + 32];
            bu1.u[0] = s1[0];
            bu1.u[1] = s1[2];

            v8f c = {0.f, 0.f, 0.f, 0.f, 0.f, 0.f, 0.f, 0.f};
            c = __builtin_amdgcn_wmma_f32_16x16x32_f16(false, a0u.v, false, bu0.v,
                                                       (short)0, c, false, false);
            c = __builtin_amdgcn_wmma_f32_16x16x32_f16(false, a1u.v, false, bu1.v,
                                                       (short)0, c, false, false);

            float xx[8] = {x0.x, x0.y, x0.z, x0.w, x1.x, x1.y, x1.z, x1.w};
            // c[r] in [1, 64e]: raw v_log_f32 (log2) is safe; log(c) = ln2 * log2(c).
            #pragma unroll
            for (int r = 0; r < 8; ++r)
                aval[r] = fmaf(LN2, __builtin_amdgcn_logf(c[r]), xx[r] + mprev);
        }

        // ---- cross-wave column max (for log-sum-exp stability) ----
        float lmax = aval[0];
        #pragma unroll
        for (int r = 1; r < 8; ++r) lmax = fmaxf(lmax, aval[r]);
        colmax[w * 32 + lane] = lmax;
        __syncthreads();

        float m = colmax[bl];
        #pragma unroll
        for (int p = 1; p < 8; ++p)
            m = fmaxf(m, colmax[(p >> 1) * 32 + (p & 1) * 16 + bl]);
        mprev = m;

        // ---- e_alpha = exp(alpha - m) packed f16 -> one ds_store_b128 ----
        // exp(a - m) = exp2(a*log2e - m*log2e); argument <= 0, raw v_exp_f32 is safe.
        const float nmL = -m * LOG2E;
        union { uint4 u; _Float16 h[8]; } pk;
        #pragma unroll
        for (int r = 0; r < 8; ++r)
            pk.h[r] = (_Float16)__builtin_amdgcn_exp2f(fmaf(aval[r], LOG2E, nmL));
        *(uint4*)&ealphaT[bl * EA_STRIDE + M0] = pk.u;

        // ---- normalizer: sum over classes of alpha at t == bs-1 (race-free slot) ----
        if (t == bs - 1) {
            float s = aval[0] + aval[1] + aval[2] + aval[3]
                    + aval[4] + aval[5] + aval[6] + aval[7];
            norm_part[(w * 2 + half_sel) * 16 + bl] = s;
        }
        __syncthreads();
    }

    if (tid < 16) {
        float s = 0.f;
        #pragma unroll
        for (int p = 0; p < 8; ++p) s += norm_part[p * 16 + tid];
        ws_norm[b0 + tid] = s;
    }
}

// Path score from pad_y, subtract normalizer, deterministic tree-reduce to scalar.
__global__ __launch_bounds__(512) void crf_score_reduce(
    const float* __restrict__ trans,
    const float* __restrict__ orig,
    const int*   __restrict__ pad_y,     // [B, T]
    const float* __restrict__ ws_norm,   // [B]
    float*       __restrict__ out)       // [1]
{
    __shared__ float red[512];
    const int i = threadIdx.x;   // one thread per batch element
    const int* y = pad_y + (size_t)i * TLEN;

    int yp = y[0];
    yp = yp < 0 ? 0 : yp;                 // TARGET_PAD_VAL = -1 -> 0
    float score = orig[yp];
    for (int t = 1; t < TLEN; ++t) {
        int yn = y[t];
        yn = yn < 0 ? 0 : yn;
        score += trans[yp * NCLS + yn];
        yp = yn;
    }
    red[i] = score - ws_norm[i];
    __syncthreads();
    for (int s = 256; s > 0; s >>= 1) {
        if (i < s) red[i] += red[i + s];
        __syncthreads();
    }
    if (i == 0) out[0] = red[0];
}

extern "C" void kernel_launch(void* const* d_in, const int* in_sizes, int n_in,
                              void* d_out, int out_size, void* d_ws, size_t ws_size,
                              hipStream_t stream) {
    (void)in_sizes; (void)n_in; (void)out_size; (void)ws_size;
    const float* pad_x = (const float*)d_in[0];   // (512, 512, 64) f32
    const float* trans = (const float*)d_in[1];   // (64, 64) f32
    const float* orig  = (const float*)d_in[2];   // (64,) f32
    const int*   pad_y = (const int*)d_in[3];     // (512, 512) i32
    const int*   bsz   = (const int*)d_in[4];     // (512,) i32
    float* ws_norm = (float*)d_ws;                // 512 floats scratch

    crf_forward_wmma<<<32, 128, 0, stream>>>(pad_x, trans, orig, bsz, ws_norm);
    crf_score_reduce<<<1, 512, 0, stream>>>(trans, orig, pad_y, ws_norm, (float*)d_out);
}
